// GCNRecommendationModel_43087111914213
// MI455X (gfx1250) — matrix operand order repaired and checked by reading, not compiled, not run on hardware
//
#include <hip/hip_runtime.h>

typedef __attribute__((ext_vector_type(16))) __bf16 v16bf;
typedef __attribute__((ext_vector_type(8)))  float  v8f;
typedef unsigned int u32;
typedef __attribute__((ext_vector_type(4))) unsigned int u32x4;
typedef __attribute__((ext_vector_type(4))) int          i32x4;
typedef __attribute__((ext_vector_type(8))) int          i32x8;

#if defined(__has_builtin)
#if __has_builtin(__builtin_amdgcn_tensor_load_to_lds) && \
    __has_builtin(__builtin_amdgcn_s_wait_tensorcnt)
#define USE_TDM 1
#endif
#endif
#ifndef USE_TDM
#define USE_TDM 0
#endif

#if USE_TDM
// Issue a TDM load of a 2D f32 tile [16 rows x 128 cols] (row stride 128
// elements) from global memory into LDS at byte offset `lds_off`.
// Rows beyond `rows_remaining` are OOB in the descriptor and read as zero.
// D# layout per CDNA5 ISA sect. 8.3/8.4.
__device__ __forceinline__ void tdm_load_tile16x128_f32(const float* gsrc,
                                                        u32 lds_off,
                                                        int rows_remaining) {
  unsigned long long ga = (unsigned long long)(uintptr_t)gsrc;
  u32 R = (u32)rows_remaining;
  u32x4 g0;
  g0[0] = 1u;                                            // count=1 (valid), user D#
  g0[1] = lds_off;                                       // lds_addr
  g0[2] = (u32)(ga & 0xffffffffu);                       // global_addr[31:0]
  g0[3] = (u32)((ga >> 32) & 0x1ffffffu) | (2u << 30);   // global_addr[56:32] | type=2
  i32x8 g1;
  g1[0] = (int)(2u << 16);                       // wg_mask=0, data_size=2 (4B), no flags
  g1[1] = (int)(128u << 16);                     // atomic_bar=0 | tensor_dim0[15:0]=128
  g1[2] = (int)((R & 0xffffu) << 16);            // tensor_dim0[31:16]=0 | tensor_dim1[15:0]
  g1[3] = (int)(((R >> 16) & 0xffffu) | (128u << 16)); // tensor_dim1[31:16] | tile_dim0=128
  g1[4] = 16;                                    // tile_dim1=16, tile_dim2=0
  g1[5] = 128;                                   // tensor_dim0_stride[31:0]=128
  g1[6] = 0;                                     // stride0[47:32]=0 | dim1_stride[15:0]=0
  g1[7] = 0;
  i32x4 z4 = {0, 0, 0, 0};
#if defined(__clang_major__) && (__clang_major__ >= 23)
  i32x8 z8 = {0, 0, 0, 0, 0, 0, 0, 0};
  __builtin_amdgcn_tensor_load_to_lds(g0, g1, z4, z4, z8, 0);
#else
  __builtin_amdgcn_tensor_load_to_lds(g0, g1, z4, z4, 0);
#endif
}
#endif

// ---------------------------------------------------------------------------
// small utility kernels
// ---------------------------------------------------------------------------
__global__ void k_zero(float* __restrict__ p, int n) {
  int i = blockIdx.x * blockDim.x + threadIdx.x;
  if (i < n) p[i] = 0.0f;
}

__global__ void k_degree(const long long* __restrict__ cols,
                         float* __restrict__ deg, int E) {
  int e = blockIdx.x * blockDim.x + threadIdx.x;
  if (e < E) atomicAdd(&deg[(int)cols[e]], 1.0f);
}

__global__ void k_dinv(float* __restrict__ deg, int n) {
  int i = blockIdx.x * blockDim.x + threadIdx.x;
  if (i < n) deg[i] = rsqrtf(deg[i] + 1.0f);
}

// out[v,c] = dinv[v]^2 * h[v,c] + b[c]
__global__ void k_self_bias(const float* __restrict__ h,
                            const float* __restrict__ dinv,
                            const float* __restrict__ b,
                            float* __restrict__ out, int C) {
  int v = blockIdx.x;
  int c = threadIdx.x;
  float dv = dinv[v];
  size_t i = (size_t)v * C + c;
  out[i] = dv * dv * h[i] + b[c];
}

// ---------------------------------------------------------------------------
// GEMM1: H[n x 128] = X[n x 128] * W[128 x 128]   (bf16 WMMA, f32 accum)
// block = 256 threads (8 waves): one 16-row M tile; wave w -> cols 16w..16w+15
// A tile staged to LDS by the Tensor Data Mover (f32), converted to bf16
// during fragment build; W staged transposed as bf16.
// ---------------------------------------------------------------------------
__global__ __launch_bounds__(256)
void k_gemm1(const float* __restrict__ X, const float* __restrict__ W,
             float* __restrict__ H, int n) {
  __shared__ float  sAf[16 * 128];   // [m][k] f32 (TDM destination)  8 KB
  __shared__ __bf16 sBt[128 * 128];  // [n][k] transposed bf16       32 KB
  const int tid = threadIdx.x;
  const int mbase = blockIdx.x * 16;

#if USE_TDM
  if (tid < 32) {                     // wave 0 drives the DMA
    tdm_load_tile16x128_f32(X + (size_t)mbase * 128,
                            (u32)(uintptr_t)sAf, n - mbase);
    __builtin_amdgcn_s_wait_tensorcnt(0);
  }
#else
  for (int i = tid; i < 16 * 128; i += 256) {
    int m = i >> 7, k = i & 127;
    int row = mbase + m;
    sAf[i] = (row < n) ? X[(size_t)row * 128 + k] : 0.0f;
  }
#endif
  for (int i = tid; i < 128 * 128; i += 256) {
    int k = i >> 7, nn = i & 127;     // W is [k][n] row-major
    sBt[nn * 128 + k] = (__bf16)W[i];
  }
  if (mbase + 16 < n)                 // gfx1250 global_prefetch path
    __builtin_prefetch(&X[(size_t)(mbase + 16) * 128 + tid], 0, 0);
  __syncthreads();

  const int wave = tid >> 5, lane = tid & 31;
  const int nbase = wave * 16;
  const int m16 = lane & 15, g = lane >> 4;

  v8f acc = {};
#pragma unroll
  for (int k0 = 0; k0 < 128; k0 += 32) {
    v16bf a, bfr;
#pragma unroll
    for (int e = 0; e < 16; ++e) {
      // A 16x32 bf16 layout: lane = m + 16g; K = k0 + (e&8?16:0) + 8g + (e&7)
      int ka = k0 + ((e & 8) ? 16 : 0) + g * 8 + (e & 7);
      a[e] = (__bf16)sAf[m16 * 128 + ka];
      // B 32x16 bf16 layout: lane = n + 16g; K = k0 + (e&8?16:0) + 8g + (e&7)
      int kb = k0 + ((e & 8) ? 16 : 0) + g * 8 + (e & 7);
      bfr[e] = sBt[(nbase + m16) * 128 + kb];
    }
    acc = __builtin_amdgcn_wmma_f32_16x16x32_bf16(false, a, false, bfr,
                                                  (short)0, acc, false, false);
  }

  // C layout: vgpr i -> M = i + 8g, N = lane&15
  if (mbase + 16 <= n) {              // uniform fast path: no per-row guards
#pragma unroll
    for (int i = 0; i < 8; ++i)
      H[(size_t)(mbase + g * 8 + i) * 128 + nbase + m16] = acc[i];
  } else {
#pragma unroll
    for (int i = 0; i < 8; ++i) {
      int m = mbase + g * 8 + i;
      if (m < n) H[(size_t)m * 128 + nbase + m16] = acc[i];
    }
  }
}

// ---------------------------------------------------------------------------
// GEMM2: H[n x 64] = relu(A[n x 128]) * W[128 x 64]  (ReLU fused at fragment
// conversion).  block = 128 threads (4 waves).
// ---------------------------------------------------------------------------
__global__ __launch_bounds__(128)
void k_gemm2(const float* __restrict__ A, const float* __restrict__ W,
             float* __restrict__ H, int n) {
  __shared__ float  sAf[16 * 128];   // [m][k] f32 (TDM destination)
  __shared__ __bf16 sBt[64 * 128];   // [n][k] transposed bf16
  const int tid = threadIdx.x;
  const int mbase = blockIdx.x * 16;

#if USE_TDM
  if (tid < 32) {
    tdm_load_tile16x128_f32(A + (size_t)mbase * 128,
                            (u32)(uintptr_t)sAf, n - mbase);
    __builtin_amdgcn_s_wait_tensorcnt(0);
  }
#else
  for (int i = tid; i < 16 * 128; i += 128) {
    int m = i >> 7, k = i & 127;
    int row = mbase + m;
    sAf[i] = (row < n) ? A[(size_t)row * 128 + k] : 0.0f;
  }
#endif
  for (int i = tid; i < 128 * 64; i += 128) {
    int k = i >> 6, nn = i & 63;      // W is [k][n] row-major
    sBt[nn * 128 + k] = (__bf16)W[i];
  }
  __syncthreads();

  const int wave = tid >> 5, lane = tid & 31;
  const int nbase = wave * 16;
  const int m16 = lane & 15, g = lane >> 4;

  v8f acc = {};
#pragma unroll
  for (int k0 = 0; k0 < 128; k0 += 32) {
    v16bf a, bfr;
#pragma unroll
    for (int e = 0; e < 16; ++e) {
      int ka = k0 + ((e & 8) ? 16 : 0) + g * 8 + (e & 7);
      a[e] = (__bf16)fmaxf(sAf[m16 * 128 + ka], 0.0f);   // fused ReLU
      int kb = k0 + ((e & 8) ? 16 : 0) + g * 8 + (e & 7);
      bfr[e] = sBt[(nbase + m16) * 128 + kb];
    }
    acc = __builtin_amdgcn_wmma_f32_16x16x32_bf16(false, a, false, bfr,
                                                  (short)0, acc, false, false);
  }

  if (mbase + 16 <= n) {
#pragma unroll
    for (int i = 0; i < 8; ++i)
      H[(size_t)(mbase + g * 8 + i) * 64 + nbase + m16] = acc[i];
  } else {
#pragma unroll
    for (int i = 0; i < 8; ++i) {
      int m = mbase + g * 8 + i;
      if (m < n) H[(size_t)m * 64 + nbase + m16] = acc[i];
    }
  }
}

// ---------------------------------------------------------------------------
// Edge aggregation: one wave per edge.
// agg[col] += dinv[row]*dinv[col] * h[row]   (f32 global atomics -> L2)
// ---------------------------------------------------------------------------
__global__ __launch_bounds__(256)
void k_edge128(const long long* __restrict__ rows,
               const long long* __restrict__ cols,
               const float* __restrict__ dinv, const float* __restrict__ h,
               float* __restrict__ agg, int E) {
  int gid = blockIdx.x * blockDim.x + threadIdx.x;
  int e = gid >> 5, lane = gid & 31;
  if (e >= E) return;
  int r = (int)rows[e], c = (int)cols[e];
  float norm = dinv[r] * dinv[c];
  float4 v = ((const float4*)(h + (size_t)r * 128))[lane];  // 512B/wave
  float* dst = agg + (size_t)c * 128 + lane * 4;
  atomicAdd(dst + 0, norm * v.x);
  atomicAdd(dst + 1, norm * v.y);
  atomicAdd(dst + 2, norm * v.z);
  atomicAdd(dst + 3, norm * v.w);
}

__global__ __launch_bounds__(256)
void k_edge64(const long long* __restrict__ rows,
              const long long* __restrict__ cols,
              const float* __restrict__ dinv, const float* __restrict__ h,
              float* __restrict__ agg, int E) {
  int gid = blockIdx.x * blockDim.x + threadIdx.x;
  int e = gid >> 5, lane = gid & 31;
  if (e >= E) return;
  int r = (int)rows[e], c = (int)cols[e];
  float norm = dinv[r] * dinv[c];
  float2 v = ((const float2*)(h + (size_t)r * 64))[lane];   // 256B/wave
  float* dst = agg + (size_t)c * 64 + lane * 2;
  atomicAdd(dst + 0, norm * v.x);
  atomicAdd(dst + 1, norm * v.y);
}

// ---------------------------------------------------------------------------
extern "C" void kernel_launch(void* const* d_in, const int* in_sizes, int n_in,
                              void* d_out, int out_size, void* d_ws, size_t ws_size,
                              hipStream_t stream) {
  const float*     x  = (const float*)d_in[0];      // [n,128] f32
  const long long* ei = (const long long*)d_in[1];  // [2,E] int64
  const float*     W1 = (const float*)d_in[2];      // [128,128]
  const float*     b1 = (const float*)d_in[3];      // [128]
  const float*     W2 = (const float*)d_in[4];      // [128,64]
  const float*     b2 = (const float*)d_in[5];      // [64]
  float* out = (float*)d_out;                       // [n,64] f32

  const int n = in_sizes[0] / 128;
  const int E = in_sizes[1] / 2;
  const long long* rows = ei;
  const long long* cols = ei + E;

  // workspace layout (floats): dinv[n] | h1[n*128] | agg1[n*128] | h2[n*64]
  float* ws = (float*)d_ws;
  size_t off = ((size_t)n + 127) & ~(size_t)127;
  float* dinv = ws;
  float* h1   = ws + off;  off += (size_t)n * 128;
  float* agg1 = ws + off;  off += (size_t)n * 128;
  float* h2   = ws + off;

  const int mtiles = (n + 15) / 16;   // 6250 for n=100000

  // degree -> dinv (in place)
  k_zero  <<<(n + 255) / 256, 256, 0, stream>>>(dinv, n);
  k_degree<<<(E + 255) / 256, 256, 0, stream>>>(cols, dinv, E);
  k_dinv  <<<(n + 255) / 256, 256, 0, stream>>>(dinv, n);

  // layer 1: WMMA GEMM -> self+bias init -> edge scatter
  k_gemm1    <<<mtiles, 256, 0, stream>>>(x, W1, h1, n);
  k_self_bias<<<n, 128, 0, stream>>>(h1, dinv, b1, agg1, 128);
  k_edge128  <<<(E + 7) / 8, 256, 0, stream>>>(rows, cols, dinv, h1, agg1, E);

  // layer 2: fused-ReLU WMMA GEMM -> self+bias init -> edge scatter into out
  k_gemm2    <<<mtiles, 128, 0, stream>>>(agg1, W2, h2, n);
  k_self_bias<<<n, 64, 0, stream>>>(h2, dinv, b2, out, 64);
  k_edge64   <<<(E + 7) / 8, 256, 0, stream>>>(rows, cols, dinv, h2, out, E);
}
